// Net_79937931313639
// MI455X (gfx1250) — matrix operand order repaired and checked by reading, not compiled, not run on hardware
//
#include <hip/hip_runtime.h>
#include <hip/hip_bf16.h>
#include <math.h>

#define KG 20
#define KN 10
#define NPTS 16384
#define EPSV 1e-8f

typedef __attribute__((ext_vector_type(16))) _Float16 v16h;
typedef __attribute__((ext_vector_type(8)))  _Float16 v8h;
typedef __attribute__((ext_vector_type(8)))  float    v8f;

// ---------------------------------------------------------------------------
// squared norms
// ---------------------------------------------------------------------------
__global__ void sqnorm_kernel(const float* __restrict__ pos, float* __restrict__ sq, int N) {
    int i = blockIdx.x * blockDim.x + threadIdx.x;
    if (i >= N) return;
    float x = pos[i * 3 + 0], y = pos[i * 3 + 1], z = pos[i * 3 + 2];
    sq[i] = x * x + y * y + z * z;
}

// ---------------------------------------------------------------------------
// brute-force KNN: one thread per query, 128-point tiles through LDS,
// per-thread top-K list kept in LDS with register-tracked worst element.
// ---------------------------------------------------------------------------
template <int K>
__global__ void knn_kernel(const float* __restrict__ pos, const float* __restrict__ sq,
                           int* __restrict__ nbr, int N) {
    __shared__ float tx[128], ty[128], tz[128], ts[128];
    __shared__ float hd[128 * K];
    __shared__ int   hi[128 * K];
    int tid = threadIdx.x;
    int q = blockIdx.x * 128 + tid;
    float px = 0.f, py = 0.f, pz = 0.f, ps = 0.f;
    if (q < N) { px = pos[q*3+0]; py = pos[q*3+1]; pz = pos[q*3+2]; ps = sq[q]; }
    float* myd = hd + tid * K;
    int*   myi = hi + tid * K;
#pragma unroll
    for (int k = 0; k < K; ++k) { myd[k] = 3.4e38f; myi[k] = 0; }
    float worst = 3.4e38f;
    int wslot = 0;
    for (int t0 = 0; t0 < N; t0 += 128) {
        __syncthreads();
        int j = t0 + tid;
        tx[tid] = pos[j*3+0]; ty[tid] = pos[j*3+1]; tz[tid] = pos[j*3+2]; ts[tid] = sq[j];
        __syncthreads();
        if (t0 + 128 < N) __builtin_prefetch(pos + (size_t)(t0 + 128) * 3, 0, 1);
        for (int jj = 0; jj < 128; ++jj) {
            float d2 = ps + ts[jj] - 2.0f * (px * tx[jj] + py * ty[jj] + pz * tz[jj]);
            if (d2 < worst) {
                myd[wslot] = d2; myi[wslot] = t0 + jj;
                worst = -3.4e38f;
#pragma unroll
                for (int k = 0; k < K; ++k) {
                    float dk = myd[k];
                    if (dk > worst) { worst = dk; wslot = k; }
                }
            }
        }
    }
    if (q < N) {
#pragma unroll
        for (int k = 0; k < K; ++k) nbr[(size_t)q * K + k] = myi[k];
    }
}

// ---------------------------------------------------------------------------
// 3x3 symmetric Jacobi eigensolver -> tangent basis (xb = largest-eig vec,
// normal = smallest-eig vec sign-fixed by orientation, yb = normal x xb)
// ---------------------------------------------------------------------------
#define JROTV(cc, ss, P, Q)                                                     \
    { float t0_, t1_;                                                           \
      t0_ = v0##P; t1_ = v0##Q; v0##P = cc*t0_ - ss*t1_; v0##Q = ss*t0_ + cc*t1_; \
      t0_ = v1##P; t1_ = v1##Q; v1##P = cc*t0_ - ss*t1_; v1##Q = ss*t0_ + cc*t1_; \
      t0_ = v2##P; t1_ = v2##Q; v2##P = cc*t0_ - ss*t1_; v2##Q = ss*t0_ + cc*t1_; }

#define JROT(APP, AQQ, APQ, ARP, ARQ, P, Q)                                     \
    { float apq_ = APQ;                                                         \
      if (fabsf(apq_) > 1e-30f) {                                               \
          float tau_ = (AQQ - APP) / (2.0f * apq_);                             \
          float t_ = (tau_ >= 0.0f ? 1.0f : -1.0f) /                            \
                     (fabsf(tau_) + sqrtf(1.0f + tau_ * tau_));                 \
          float c_ = rsqrtf(1.0f + t_ * t_);                                    \
          float s_ = t_ * c_;                                                   \
          APP -= t_ * apq_; AQQ += t_ * apq_; APQ = 0.0f;                       \
          float rp_ = ARP, rq_ = ARQ;                                           \
          ARP = c_ * rp_ - s_ * rq_; ARQ = s_ * rp_ + c_ * rq_;                 \
          JROTV(c_, s_, P, Q)                                                   \
      } }

__global__ void basis_kernel(const float* __restrict__ pos, const int* __restrict__ nbr,
                             float* __restrict__ xb, float* __restrict__ yb, int N) {
    int n = blockIdx.x * blockDim.x + threadIdx.x;
    if (n >= N) return;
    float px = pos[n*3+0], py = pos[n*3+1], pz = pos[n*3+2];
    float a00 = 0.f, a01 = 0.f, a02 = 0.f, a11 = 0.f, a12 = 0.f, a22 = 0.f;
#pragma unroll
    for (int k = 0; k < KN; ++k) {
        int j = nbr[(size_t)n * KN + k];
        float rx = pos[j*3+0] - px, ry = pos[j*3+1] - py, rz = pos[j*3+2] - pz;
        a00 += rx * rx; a01 += rx * ry; a02 += rx * rz;
        a11 += ry * ry; a12 += ry * rz; a22 += rz * rz;
    }
    float v00 = 1.f, v01 = 0.f, v02 = 0.f;
    float v10 = 0.f, v11 = 1.f, v12 = 0.f;
    float v20 = 0.f, v21 = 0.f, v22 = 1.f;
    for (int sweep = 0; sweep < 8; ++sweep) {
        JROT(a00, a11, a01, a02, a12, 0, 1)
        JROT(a00, a22, a02, a01, a12, 0, 2)
        JROT(a11, a22, a12, a01, a02, 1, 2)
    }
    int imin = 0; float emin = a00;
    if (a11 < emin) { emin = a11; imin = 1; }
    if (a22 < emin) { emin = a22; imin = 2; }
    int imax = 0; float emax = a00;
    if (a11 > emax) { emax = a11; imax = 1; }
    if (a22 > emax) { emax = a22; imax = 2; }
    if (imax == imin) imax = (imin + 1) % 3;
    float nx = (imin == 0) ? v00 : ((imin == 1) ? v01 : v02);
    float ny = (imin == 0) ? v10 : ((imin == 1) ? v11 : v12);
    float nz = (imin == 0) ? v20 : ((imin == 1) ? v21 : v22);
    float xx = (imax == 0) ? v00 : ((imax == 1) ? v01 : v02);
    float xy = (imax == 0) ? v10 : ((imax == 1) ? v11 : v12);
    float xz = (imax == 0) ? v20 : ((imax == 1) ? v21 : v22);
    float sgn = (nx * px + ny * py + nz * pz) < 0.0f ? -1.0f : 1.0f;
    nx *= sgn; ny *= sgn; nz *= sgn;
    float yx = ny * xz - nz * xy;
    float yy = nz * xx - nx * xz;
    float yz = nx * xy - ny * xx;
    xb[n*3+0] = xx; xb[n*3+1] = xy; xb[n*3+2] = xz;
    yb[n*3+0] = yx; yb[n*3+1] = yy; yb[n*3+2] = yz;
}

// ---------------------------------------------------------------------------
// moving-least-squares gradient operators Gx, Gy and 2x2 frame rotations R
// ---------------------------------------------------------------------------
__global__ void graddiv_kernel(const float* __restrict__ pos,
                               const float* __restrict__ xb, const float* __restrict__ yb,
                               const int* __restrict__ nbr,
                               float* __restrict__ Gx, float* __restrict__ Gy,
                               float* __restrict__ R, int N) {
    int n = blockIdx.x * blockDim.x + threadIdx.x;
    if (n >= N) return;
    float px = pos[n*3+0], py = pos[n*3+1], pz = pos[n*3+2];
    float xbx = xb[n*3+0], xby = xb[n*3+1], xbz = xb[n*3+2];
    float ybx = yb[n*3+0], yby = yb[n*3+1], ybz = yb[n*3+2];
    // pass 1: bandwidth h = mean distance
    float dsum = 0.f;
#pragma unroll
    for (int k = 0; k < KG; ++k) {
        int j = nbr[(size_t)n * KG + k];
        float rx = pos[j*3+0] - px, ry = pos[j*3+1] - py, rz = pos[j*3+2] - pz;
        dsum += sqrtf(rx * rx + ry * ry + rz * rz + EPSV);
    }
    float h = dsum * (1.0f / KG);
    float denom = h * h + EPSV;
    // pass 2: normal equations A = B^T W B + reg
    float A00 = 0.f, A01 = 0.f, A02 = 0.f, A11 = 0.f, A12 = 0.f, A22 = 0.f;
#pragma unroll
    for (int k = 0; k < KG; ++k) {
        int j = nbr[(size_t)n * KG + k];
        float rx = pos[j*3+0] - px, ry = pos[j*3+1] - py, rz = pos[j*3+2] - pz;
        float d2 = rx * rx + ry * ry + rz * rz + EPSV;
        float w = __expf(-d2 / denom);
        float u = rx * xbx + ry * xby + rz * xbz;
        float v = rx * ybx + ry * yby + rz * ybz;
        A00 += w; A01 += w * u; A02 += w * v;
        A11 += w * u * u; A12 += w * u * v; A22 += w * v * v;
    }
    A00 += 0.001f; A11 += 0.001f; A22 += 0.001f;
    float det = A00 * (A11 * A22 - A12 * A12)
              - A01 * (A01 * A22 - A12 * A02)
              + A02 * (A01 * A12 - A11 * A02);
    float inv = 1.0f / det;
    float i10 = (A02 * A12 - A01 * A22) * inv;
    float i11 = (A00 * A22 - A02 * A02) * inv;
    float i12 = (A01 * A02 - A00 * A12) * inv;
    float i20 = (A01 * A12 - A02 * A11) * inv;
    float i21 = i12;
    float i22 = (A00 * A11 - A01 * A01) * inv;
    // pass 3: per-neighbor operator weights + frame rotation
#pragma unroll
    for (int k = 0; k < KG; ++k) {
        int j = nbr[(size_t)n * KG + k];
        float rx = pos[j*3+0] - px, ry = pos[j*3+1] - py, rz = pos[j*3+2] - pz;
        float d2 = rx * rx + ry * ry + rz * rz + EPSV;
        float w = __expf(-d2 / denom);
        float u = rx * xbx + ry * xby + rz * xbz;
        float v = rx * ybx + ry * yby + rz * ybz;
        Gx[(size_t)n * KG + k] = w * (i10 + i11 * u + i12 * v);
        Gy[(size_t)n * KG + k] = w * (i20 + i21 * u + i22 * v);
        float xjx = xb[j*3+0], xjy = xb[j*3+1], xjz = xb[j*3+2];
        float yjx = yb[j*3+0], yjy = yb[j*3+1], yjz = yb[j*3+2];
        size_t rb = ((size_t)n * KG + k) * 4;
        R[rb + 0] = xbx * xjx + xby * xjy + xbz * xjz;  // R00 = xb_n . xb_j
        R[rb + 1] = xbx * yjx + xby * yjy + xbz * yjz;  // R01 = xb_n . yb_j
        R[rb + 2] = ybx * xjx + yby * xjy + ybz * xjz;  // R10
        R[rb + 3] = ybx * yjx + yby * yjy + ybz * yjz;  // R11
    }
}

// initial vector field: v = grad(pos)  (C = 3)
__global__ void grad_pos_kernel(const float* __restrict__ pos, const int* __restrict__ nbr,
                                const float* __restrict__ Gx, const float* __restrict__ Gy,
                                float* __restrict__ v, int N) {
    int n = blockIdx.x * blockDim.x + threadIdx.x;
    if (n >= N) return;
    float ax0 = 0.f, ax1 = 0.f, ax2 = 0.f, ay0 = 0.f, ay1 = 0.f, ay2 = 0.f;
#pragma unroll
    for (int k = 0; k < KG; ++k) {
        int j = nbr[(size_t)n * KG + k];
        float gx = Gx[(size_t)n * KG + k], gy = Gy[(size_t)n * KG + k];
        float p0 = pos[j*3+0], p1 = pos[j*3+1], p2 = pos[j*3+2];
        ax0 += gx * p0; ax1 += gx * p1; ax2 += gx * p2;
        ay0 += gy * p0; ay1 += gy * p1; ay2 += gy * p2;
    }
    v[((size_t)n*2+0)*3+0] = ax0; v[((size_t)n*2+0)*3+1] = ax1; v[((size_t)n*2+0)*3+2] = ax2;
    v[((size_t)n*2+1)*3+0] = ay0; v[((size_t)n*2+1)*3+1] = ay1; v[((size_t)n*2+1)*3+2] = ay2;
}

// ---------------------------------------------------------------------------
// weight convert: f32 (rows x cols) -> f16 transposed (cols x rowsPad), zero pad
// ---------------------------------------------------------------------------
__global__ void convert_wT_kernel(const float* __restrict__ W, _Float16* __restrict__ WT,
                                  int rows, int cols, int rowsPad) {
    int idx = blockIdx.x * blockDim.x + threadIdx.x;
    if (idx >= cols * rowsPad) return;
    int c = idx / rowsPad;
    int r = idx % rowsPad;
    WT[(size_t)c * rowsPad + r] = (r < rows) ? (_Float16)W[(size_t)r * cols + c] : (_Float16)0.f;
}

// ---------------------------------------------------------------------------
// per-point features: [max-pool x[nbr] (opt. centralized) | div(v) | curl(v)]
// written as f16, K-padded for WMMA
// ---------------------------------------------------------------------------
__global__ void features_kernel(const float* __restrict__ x, const float* __restrict__ v,
                                const int* __restrict__ nbr,
                                const float* __restrict__ Gx, const float* __restrict__ Gy,
                                const float* __restrict__ R,
                                _Float16* __restrict__ F,
                                int cin, int Kpad, int centralized) {
    int n = blockIdx.x;
    for (int c = threadIdx.x; c < cin; c += blockDim.x) {
        float xc = x[(size_t)n * cin + c];
        float xmax = -3.4e38f, dv = 0.f, cv = 0.f;
#pragma unroll
        for (int k = 0; k < KG; ++k) {
            int j = nbr[(size_t)n * KG + k];
            float xn = x[(size_t)j * cin + c];
            if (centralized) xn -= xc;
            xmax = fmaxf(xmax, xn);
            float gx = Gx[(size_t)n * KG + k], gy = Gy[(size_t)n * KG + k];
            size_t rb = ((size_t)n * KG + k) * 4;
            float r00 = R[rb+0], r01 = R[rb+1], r10 = R[rb+2], r11 = R[rb+3];
            float v0 = v[((size_t)j*2+0)*cin + c];
            float v1 = v[((size_t)j*2+1)*cin + c];
            dv += gx * (r00 * v0 + r01 * v1) + gy * (r10 * v0 + r11 * v1);
            // Jv[nbr] = (-v1, v0)
            cv += gx * (r01 * v0 - r00 * v1) + gy * (r11 * v0 - r10 * v1);
        }
        size_t fb = (size_t)n * Kpad;
        F[fb + c]            = (_Float16)xmax;
        F[fb + cin + c]      = (_Float16)dv;
        F[fb + 2 * cin + c]  = (_Float16)cv;
    }
    for (int c = 3 * cin + threadIdx.x; c < Kpad; c += blockDim.x)
        F[(size_t)n * Kpad + c] = (_Float16)0.f;
}

// ---------------------------------------------------------------------------
// WMMA GEMM: D(MxNcols,f32) = relu?(A(MxK,f16 row-major) * BT(NcolsxK,f16)^T + bias)
// one wave per 16x16 output tile, K-loop in steps of 32 via v_wmma_f32_16x16x32_f16
// ---------------------------------------------------------------------------
__global__ void wmma_gemm_kernel(const _Float16* __restrict__ A, const _Float16* __restrict__ BT,
                                 const float* __restrict__ bias, float* __restrict__ D,
                                 int Kdim, int Ncols, int relu) {
    int lane = threadIdx.x & 31;
    int rowBase = blockIdx.x * 16;
    int colBase = blockIdx.y * 16;
    int l15 = lane & 15;
    int half = lane >> 4;
    const _Float16* ap = A  + (size_t)(rowBase + l15) * Kdim + half * 16;
    const _Float16* bp = BT + (size_t)(colBase + l15) * Kdim + half * 16;
    v8f acc = {};
    for (int kb = 0; kb < Kdim; kb += 32) {
        // A 16x32 f16 layout: lanes 0-15 hold K 0..15; lanes 16-31 hold K 16..31
        v8h a0 = *(const v8h*)(ap + kb);
        v8h a1 = *(const v8h*)(ap + kb + 8);
        v16h a = __builtin_shufflevector(a0, a1, 0,1,2,3,4,5,6,7,8,9,10,11,12,13,14,15);
        // B 32x16 f16: lane = column, lanes 0-15 K 0..15, lanes 16-31 K 16..31
        v16h b = *(const v16h*)(bp + kb);
        acc = __builtin_amdgcn_wmma_f32_16x16x32_f16(false, a, false, b, (short)0, acc,
                                                     false, false);
    }
    float bcol = bias ? bias[colBase + l15] : 0.0f;
#pragma unroll
    for (int r = 0; r < 8; ++r) {
        int m = rowBase + half * 8 + r;            // C/D layout: vgpr r -> M = r (+8 for upper half)
        float val = acc[r] + bcol;
        if (relu) val = fmaxf(val, 0.0f);
        D[(size_t)m * Ncols + colBase + l15] = val;
    }
}

// ---------------------------------------------------------------------------
// gv = grad(x_out); pack [v | gv] rows (2N x Kvpad) as f16 for the Wv GEMM
// ---------------------------------------------------------------------------
__global__ void gv_pack_kernel(const float* __restrict__ v, const float* __restrict__ xout,
                               const int* __restrict__ nbr,
                               const float* __restrict__ Gx, const float* __restrict__ Gy,
                               _Float16* __restrict__ Vin,
                               int cin, int cout, int Kvpad) {
    int n = blockIdx.x;
    size_t r0 = (size_t)(2 * n + 0) * Kvpad;
    size_t r1 = (size_t)(2 * n + 1) * Kvpad;
    for (int c = threadIdx.x; c < cin; c += blockDim.x) {
        Vin[r0 + c] = (_Float16)v[((size_t)n*2+0)*cin + c];
        Vin[r1 + c] = (_Float16)v[((size_t)n*2+1)*cin + c];
    }
    for (int c = threadIdx.x; c < cout; c += blockDim.x) {
        float gx = 0.f, gy = 0.f;
#pragma unroll
        for (int k = 0; k < KG; ++k) {
            int j = nbr[(size_t)n * KG + k];
            float f = xout[(size_t)j * cout + c];
            gx += Gx[(size_t)n * KG + k] * f;
            gy += Gy[(size_t)n * KG + k] * f;
        }
        Vin[r0 + cin + c] = (_Float16)gx;
        Vin[r1 + cin + c] = (_Float16)gy;
    }
    for (int c = cin + cout + threadIdx.x; c < Kvpad; c += blockDim.x) {
        Vin[r0 + c] = (_Float16)0.f;
        Vin[r1 + c] = (_Float16)0.f;
    }
}

// vector nonlinearity: vo *= relu(|vo| + bv) / (|vo| + eps), norm over the 2 components
__global__ void vnorm_kernel(const float* __restrict__ vo, const float* __restrict__ bv,
                             float* __restrict__ vnext, int N, int cout) {
    int idx = blockIdx.x * blockDim.x + threadIdx.x;
    if (idx >= N * cout) return;
    int n = idx / cout, c = idx % cout;
    float a0 = vo[((size_t)n*2+0)*cout + c];
    float a1 = vo[((size_t)n*2+1)*cout + c];
    float nn = sqrtf(a0 * a0 + a1 * a1 + EPSV);
    float s = fmaxf(nn + bv[c], 0.0f) / (nn + EPSV);
    vnext[((size_t)n*2+0)*cout + c] = a0 * s;
    vnext[((size_t)n*2+1)*cout + c] = a1 * s;
}

// ---------------------------------------------------------------------------
// host launch
// ---------------------------------------------------------------------------
extern "C" void kernel_launch(void* const* d_in, const int* in_sizes, int n_in,
                              void* d_out, int out_size, void* d_ws, size_t ws_size,
                              hipStream_t stream) {
    const int N = NPTS;
    const float* points = (const float*)d_in[0];
    const float* als    = (const float*)d_in[1];
    const float* Ws0 = (const float*)d_in[2];
    const float* bs0 = (const float*)d_in[3];
    const float* Wv0 = (const float*)d_in[4];
    const float* bv0 = (const float*)d_in[5];
    const float* Ws1 = (const float*)d_in[6];
    const float* bs1 = (const float*)d_in[7];
    const float* Wv1 = (const float*)d_in[8];
    const float* bv1 = (const float*)d_in[9];
    const float* Ws2 = (const float*)d_in[10];
    const float* bs2 = (const float*)d_in[11];

    char* ws = (char*)d_ws;
    size_t off = 0;
    auto alloc = [&](size_t bytes) -> void* {
        void* p = ws + off;
        off = (off + bytes + 255) & ~(size_t)255;
        return p;
    };

    float* pos   = (float*)alloc((size_t)N * 3 * 4);
    float* sq    = (float*)alloc((size_t)N * 4);
    int*   nbr20 = (int*)  alloc((size_t)N * KG * 4);
    int*   nbr10 = (int*)  alloc((size_t)N * KN * 4);
    float* xb    = (float*)alloc((size_t)N * 3 * 4);
    float* yb    = (float*)alloc((size_t)N * 3 * 4);
    float* Gx    = (float*)alloc((size_t)N * KG * 4);
    float* Gy    = (float*)alloc((size_t)N * KG * 4);
    float* Rm    = (float*)alloc((size_t)N * KG * 4 * 4);
    float* vA    = (float*)alloc((size_t)N * 2 * 128 * 4);
    float* vB    = (float*)alloc((size_t)N * 2 * 128 * 4);
    float* vo    = (float*)alloc((size_t)N * 2 * 128 * 4);
    _Float16* F    = (_Float16*)alloc((size_t)N * 384 * 2);
    _Float16* Vin  = (_Float16*)alloc((size_t)N * 2 * 192 * 2);
    _Float16* ws0T = (_Float16*)alloc((size_t)64 * 32 * 2);
    _Float16* wv0T = (_Float16*)alloc((size_t)64 * 96 * 2);
    _Float16* ws1T = (_Float16*)alloc((size_t)128 * 192 * 2);
    _Float16* wv1T = (_Float16*)alloc((size_t)128 * 192 * 2);
    _Float16* ws2T = (_Float16*)alloc((size_t)256 * 384 * 2);

    float* out0 = (float*)d_out;                 // N x 64
    float* out1 = out0 + (size_t)N * 64;         // N x 128
    float* out2 = out1 + (size_t)N * 128;        // N x 256

    // pos = concat(points, als_ppoints)
    hipMemcpyAsync(pos, points, (size_t)8192 * 3 * 4, hipMemcpyDeviceToDevice, stream);
    hipMemcpyAsync(pos + (size_t)8192 * 3, als, (size_t)8192 * 3 * 4,
                   hipMemcpyDeviceToDevice, stream);

    sqnorm_kernel<<<(N + 255) / 256, 256, 0, stream>>>(pos, sq, N);
    knn_kernel<KG><<<N / 128, 128, 0, stream>>>(pos, sq, nbr20, N);
    knn_kernel<KN><<<N / 128, 128, 0, stream>>>(pos, sq, nbr10, N);
    basis_kernel<<<(N + 255) / 256, 256, 0, stream>>>(pos, nbr10, xb, yb, N);
    graddiv_kernel<<<(N + 255) / 256, 256, 0, stream>>>(pos, xb, yb, nbr20, Gx, Gy, Rm, N);
    grad_pos_kernel<<<(N + 255) / 256, 256, 0, stream>>>(pos, nbr20, Gx, Gy, vA, N);

    // weights -> transposed f16 with K padding
    convert_wT_kernel<<<(64 * 32 + 255) / 256, 256, 0, stream>>>(Ws0, ws0T, 9, 64, 32);
    convert_wT_kernel<<<(64 * 96 + 255) / 256, 256, 0, stream>>>(Wv0, wv0T, 67, 64, 96);
    convert_wT_kernel<<<(128 * 192 + 255) / 256, 256, 0, stream>>>(Ws1, ws1T, 192, 128, 192);
    convert_wT_kernel<<<(128 * 192 + 255) / 256, 256, 0, stream>>>(Wv1, wv1T, 192, 128, 192);
    convert_wT_kernel<<<(256 * 384 + 255) / 256, 256, 0, stream>>>(Ws2, ws2T, 384, 256, 384);

    // ---- layer 0: cin=3 -> cout=64, centralized max-pool ----
    features_kernel<<<N, 64, 0, stream>>>(pos, vA, nbr20, Gx, Gy, Rm, F, 3, 32, 1);
    wmma_gemm_kernel<<<dim3(N / 16, 64 / 16), 32, 0, stream>>>(F, ws0T, bs0, out0, 32, 64, 1);
    gv_pack_kernel<<<N, 64, 0, stream>>>(vA, out0, nbr20, Gx, Gy, Vin, 3, 64, 96);
    wmma_gemm_kernel<<<dim3(2 * N / 16, 64 / 16), 32, 0, stream>>>(Vin, wv0T, nullptr, vo, 96, 64, 0);
    vnorm_kernel<<<(N * 64 + 255) / 256, 256, 0, stream>>>(vo, bv0, vB, N, 64);

    // ---- layer 1: cin=64 -> cout=128 ----
    features_kernel<<<N, 64, 0, stream>>>(out0, vB, nbr20, Gx, Gy, Rm, F, 64, 192, 0);
    wmma_gemm_kernel<<<dim3(N / 16, 128 / 16), 32, 0, stream>>>(F, ws1T, bs1, out1, 192, 128, 1);
    gv_pack_kernel<<<N, 128, 0, stream>>>(vB, out1, nbr20, Gx, Gy, Vin, 64, 128, 192);
    wmma_gemm_kernel<<<dim3(2 * N / 16, 128 / 16), 32, 0, stream>>>(Vin, wv1T, nullptr, vo, 192, 128, 0);
    vnorm_kernel<<<(N * 128 + 255) / 256, 256, 0, stream>>>(vo, bv1, vA, N, 128);

    // ---- layer 2: cin=128 -> cout=256 (no vector update) ----
    features_kernel<<<N, 128, 0, stream>>>(out1, vA, nbr20, Gx, Gy, Rm, F, 128, 384, 0);
    wmma_gemm_kernel<<<dim3(N / 16, 256 / 16), 32, 0, stream>>>(F, ws2T, bs2, out2, 384, 256, 1);
}
